// CARCAttentionProcessor_29807073034446
// MI455X (gfx1250) — matrix operand order repaired and checked by reading, not compiled, not run on hardware
//
#include <hip/hip_runtime.h>

// ---------------------------------------------------------------------------
// CARC attention processor for MI455X (gfx1250, wave32, WMMA f16 16x16x32).
// qkv_gemm (128x128 tiles) -> bank_pool -> attn (flash, 64-key chunks with
// double-buffered async global->LDS K/V staging) -> out_gemm (128x128 tiles).
// ---------------------------------------------------------------------------

typedef __attribute__((ext_vector_type(16))) _Float16 v16h;
typedef __attribute__((ext_vector_type(8)))  _Float16 v8h;
typedef __attribute__((ext_vector_type(4)))  _Float16 v4h;
typedef __attribute__((ext_vector_type(2)))  _Float16 v2h;
typedef __attribute__((ext_vector_type(8)))  float    v8f;
typedef __attribute__((ext_vector_type(4)))  float    v4f;

#define DEV __device__ __forceinline__

DEV v8f wmma_f32_f16(v16h a, v16h b, v8f c) {
    return __builtin_amdgcn_wmma_f32_16x16x32_f16(
        false, a, false, b, (short)0, c, false, false);
}

// A-matrix fragment (16x32 f16): lane holds halves [b1..b1+7],[b1+16..b1+23].
DEV v16h ld_frag_gap(const _Float16* p) {
    union { v16h v; v8h h[2]; } u;
    u.h[0] = *(const v8h*)(p);
    u.h[1] = *(const v8h*)(p + 16);
    return u.v;
}
// Contiguous-k fragment (this code's order, used for A and B): lane holds 16 consecutive K halves.
DEV v16h ld_frag_cont(const _Float16* p) {
    union { v16h v; v8h h[2]; } u;
    u.h[0] = *(const v8h*)(p);
    u.h[1] = *(const v8h*)(p + 8);
    return u.v;
}

// Generic shared pointer -> 32-bit LDS byte address (low 32 bits of the
// shared-aperture generic address are the LDS offset per the CDNA5 ISA).
DEV unsigned lds_addr(const void* p) { return (unsigned)(size_t)p; }

// Async 16B global->LDS copy per lane (ASYNCcnt-tracked, gfx1250).
DEV void async_b128(unsigned lds, const void* g) {
    asm volatile("global_load_async_to_lds_b128 %0, %1, off"
                 :: "v"(lds), "v"(g) : "memory");
}
DEV void wait_async0() {
#if __has_builtin(__builtin_amdgcn_s_wait_asynccnt)
    __builtin_amdgcn_s_wait_asynccnt(0);
#else
    asm volatile("s_wait_asynccnt 0x0" ::: "memory");
#endif
}

// Constants: B=4, L=1024, C=1280, H=20, Dh=64, BH=80, keys=1024+256=1280
// ALPHA = 0.48 ; score scale 1/sqrt(64)=0.125 folded into Q.

// ---------------------------------------------------------------------------
// Kernel 1: QKV projections.  M=4096, N=1280, K=1280.  Block = 8 waves,
// 128(M) x 128(N) tile; wave w owns rows [16w,16w+16) -> 8 WMMAs per K-step.
// ---------------------------------------------------------------------------
__global__ __launch_bounds__(256) void qkv_gemm(
    const float* __restrict__ X,
    const float* __restrict__ Wq, const float* __restrict__ Wk,
    const float* __restrict__ Wv,
    _Float16* __restrict__ Qs,   // [80][1024][64], pre-scaled by 0.125
    _Float16* __restrict__ Kc,   // [80][1280][64] (tokens 0..1023)
    _Float16* __restrict__ Vt)   // [80][64][1280] (tokens 0..1023), transposed
{
    __shared__ __align__(16) _Float16 At[128 * 32];   // 8 KB
    __shared__ __align__(16) _Float16 Bt[8][16 * 32]; // 8 KB

    const int tid  = threadIdx.x;
    const int wave = tid >> 5;
    const int lane = tid & 31;
    const int m0 = blockIdx.x * 128;
    const int n0 = blockIdx.y * 128;
    const int z  = blockIdx.z;
    const float* W = (z == 0) ? Wq : (z == 1) ? Wk : Wv;

    const int row = lane & 15;
    const int b1  = (lane < 16) ? 0 : 8;
    const int bb  = (lane < 16) ? 0 : 16;

    v8f acc[8] = {};
    for (int kk = 0; kk < 1280; kk += 32) {
        if (kk + 32 < 1280) {
            __builtin_prefetch(&X[(m0 + (tid >> 1)) * 1280 + kk + 32], 0, 0);
            __builtin_prefetch(&W[(kk + 32 + (tid >> 3)) * 1280 + n0], 0, 0);
        }
        // stage A tile 128x32 (float4 loads, b64 LDS stores)
        #pragma unroll
        for (int i = 0; i < 4; ++i) {
            int idx4 = tid + i * 256;
            int r = idx4 >> 3;
            int c = (idx4 & 7) * 4;
            v4f x = *(const v4f*)&X[(m0 + r) * 1280 + kk + c];
            v4h h; h.x = (_Float16)x.x; h.y = (_Float16)x.y;
            h.z = (_Float16)x.z; h.w = (_Float16)x.w;
            *(v4h*)&At[r * 32 + c] = h;
        }
        // stage W tile 32(K)x128(N) transposed; pack 2 K-rows -> b32 stores
        #pragma unroll
        for (int i = 0; i < 2; ++i) {
            int idx = tid + i * 256;          // 512 items: 16 kpairs x 32 quads
            int kp = idx >> 5;                // K pair 0..15
            int c  = (idx & 31) * 4;          // N col 0..124
            v4f x0 = *(const v4f*)&W[(kk + 2 * kp    ) * 1280 + n0 + c];
            v4f x1 = *(const v4f*)&W[(kk + 2 * kp + 1) * 1280 + n0 + c];
            #pragma unroll
            for (int u = 0; u < 4; ++u) {
                int cc = c + u;
                v2h h; h.x = (_Float16)x0[u]; h.y = (_Float16)x1[u];
                *(v2h*)&Bt[cc >> 4][(cc & 15) * 32 + 2 * kp] = h;
            }
        }
        __syncthreads();
        v16h a = ld_frag_cont(&At[(wave * 16 + row) * 32 + bb]);
        #pragma unroll
        for (int nt = 0; nt < 8; ++nt) {
            v16h b = ld_frag_cont(&Bt[nt][row * 32 + bb]);
            acc[nt] = wmma_f32_f16(a, b, acc[nt]);
        }
        __syncthreads();
    }

    #pragma unroll
    for (int nt = 0; nt < 8; ++nt) {
        #pragma unroll
        for (int r = 0; r < 8; ++r) {
            int m = m0 + wave * 16 + r + ((lane < 16) ? 0 : 8);
            int n = n0 + nt * 16 + (lane & 15);
            int bidx = m >> 10, l = m & 1023;
            int h = n >> 6,     d = n & 63;
            int bh = bidx * 20 + h;
            float v = acc[nt][r];
            if (z == 0)      Qs[(bh * 1024 + l) * 64 + d] = (_Float16)(v * 0.125f);
            else if (z == 1) Kc[(bh * 1280 + l) * 64 + d] = (_Float16)v;
            else             Vt[(bh * 64 + d) * 1280 + l] = (_Float16)v;
        }
    }
}

// ---------------------------------------------------------------------------
// Kernel 2: bank fp16-round + 2x2 avg-pool + ALPHA into tokens 1024..1279.
// ---------------------------------------------------------------------------
__global__ __launch_bounds__(256) void bank_pool(
    const float* __restrict__ Kbg, const float* __restrict__ Vbg,
    _Float16* __restrict__ Kc, _Float16* __restrict__ Vt)
{
    int idx = blockIdx.x * 256 + threadIdx.x;   // 40*256*64 total
    int d = idx & 63;
    int j = (idx >> 6) & 255;
    int g = idx >> 14;
    int gy = j >> 4, gx = j & 15;
    int t00 = (2 * gy) * 32 + 2 * gx;
    const float* kp = Kbg + (g * 1024) * 64 + d;
    const float* vp = Vbg + (g * 1024) * 64 + d;
    float ks = 0.f, vs = 0.f;
    #pragma unroll
    for (int dy = 0; dy < 2; ++dy)
        #pragma unroll
        for (int dx = 0; dx < 2; ++dx) {
            int tok = t00 + dy * 32 + dx;
            ks += (float)(_Float16)kp[tok * 64];
            vs += (float)(_Float16)vp[tok * 64];
        }
    _Float16 kh = (_Float16)(ks * 0.25f * 0.48f);
    _Float16 vh = (_Float16)(vs * 0.25f * 0.48f);
    int tokd = 1024 + j;
    #pragma unroll
    for (int rep = 0; rep < 2; ++rep) {
        int bh = g + rep * 40;
        Kc[(bh * 1280 + tokd) * 64 + d] = kh;
        Vt[(bh * 64 + d) * 1280 + tokd] = vh;
    }
}

// ---------------------------------------------------------------------------
// Kernel 3: flash attention, 64-key chunks.  grid = (1024/128, 80); 8 waves.
// K/V chunks staged once per block into LDS via async global->LDS copies,
// double-buffered; all 8 waves consume the same chunk.  Per chunk: 8 score
// WMMAs, online softmax, P via per-wave LDS, 8 PV WMMAs.
// ---------------------------------------------------------------------------
__global__ __launch_bounds__(256) void attn(
    const _Float16* __restrict__ Qs, const _Float16* __restrict__ Kc,
    const _Float16* __restrict__ Vt, _Float16* __restrict__ ctxh)
{
    __shared__ __align__(16) _Float16 Plds[8][16 * 64];  // 16 KB
    __shared__ __align__(16) _Float16 Kld[2][64 * 64];   // 16 KB (2 x 8 KB)
    __shared__ __align__(16) _Float16 Vld[2][64 * 64];   // 16 KB

    const int tid  = threadIdx.x;
    const int wave = tid >> 5;
    const int lane = tid & 31;
    const int bh = blockIdx.y;
    const int q0 = blockIdx.x * 128 + wave * 16;

    const int row = lane & 15;
    const int b1  = (lane < 16) ? 0 : 8;
    const int bb  = (lane < 16) ? 0 : 16;

    const _Float16* qrow = Qs + (size_t)(bh * 1024 + q0 + row) * 64;
    v16h qa0 = ld_frag_cont(qrow + 0  + bb);
    v16h qa1 = ld_frag_cont(qrow + 32 + bb);

    float mrun[8], lrun[8];
    v8f oacc[4] = {{}, {}, {}, {}};
    #pragma unroll
    for (int r = 0; r < 8; ++r) { mrun[r] = -1e30f; lrun[r] = 0.f; }

    const _Float16* kbase = Kc + (size_t)bh * 1280 * 64;
    const _Float16* vbase = Vt + (size_t)bh * 64 * 1280;
    _Float16* pw = &Plds[wave][0];

    // Issue async copies of one 64-key chunk (8 KB K + 8 KB V) into buf.
    // Block-wide: 512 x 16B per tensor; this wave contributes 2+2 copies.
    auto issue = [&](int t0, int buf) {
        unsigned kl = lds_addr(&Kld[buf][0]);
        unsigned vl = lds_addr(&Vld[buf][0]);
        #pragma unroll
        for (int i = 0; i < 2; ++i) {
            int idx = wave * 64 + i * 32 + lane;          // 0..511
            // K chunk: contiguous 8 KB at kbase + t0*64 halves
            async_b128(kl + idx * 16, kbase + (size_t)t0 * 64 + idx * 8);
            // V chunk: 64 d-rows of 128B, global row stride 2560B
            int dr = idx >> 3;
            int dc = (idx & 7) * 8;
            async_b128(vl + idx * 16, vbase + (size_t)dr * 1280 + t0 + dc);
        }
    };

    issue(0, 0);
    for (int it = 0; it < 20; ++it) {
        const int t0  = it * 64;
        const int cur = it & 1;
        wait_async0();        // this wave's copies for chunk `it` landed
        __syncthreads();      // everyone's copies landed; prev compute done
        if (it + 1 < 20) issue(t0 + 64, cur ^ 1);   // overlap with compute

        const _Float16* kl = &Kld[cur][0];
        const _Float16* vl = &Vld[cur][0];

        // ---- scores: four 16-key subtiles over d=0..63 ----
        v8f s[4];
        #pragma unroll
        for (int st = 0; st < 4; ++st) {
            const _Float16* kr = kl + (st * 16 + row) * 64 + bb;
            v8f t = {};
            t = wmma_f32_f16(qa0, ld_frag_cont(kr),      t);
            t = wmma_f32_f16(qa1, ld_frag_cont(kr + 32), t);
            s[st] = t;
        }

        // ---- online softmax per row ----
        #pragma unroll
        for (int r = 0; r < 8; ++r) {
            float mx = fmaxf(fmaxf(s[0][r], s[1][r]), fmaxf(s[2][r], s[3][r]));
            #pragma unroll
            for (int sh = 1; sh < 16; sh <<= 1)
                mx = fmaxf(mx, __shfl_xor(mx, sh, 32));
            float mnew = fmaxf(mrun[r], mx);
            float corr = __expf(mrun[r] - mnew);
            float p[4], ps = 0.f;
            #pragma unroll
            for (int st = 0; st < 4; ++st) { p[st] = __expf(s[st][r] - mnew); ps += p[st]; }
            #pragma unroll
            for (int sh = 1; sh < 16; sh <<= 1)
                ps += __shfl_xor(ps, sh, 32);
            lrun[r] = lrun[r] * corr + ps;
            mrun[r] = mnew;
            #pragma unroll
            for (int dt = 0; dt < 4; ++dt) oacc[dt][r] *= corr;
            int rr = r + ((lane < 16) ? 0 : 8);
            #pragma unroll
            for (int st = 0; st < 4; ++st)
                pw[rr * 64 + st * 16 + (lane & 15)] = (_Float16)p[st];
        }

        // ---- P (16x64) as two A fragments; V from LDS as B fragments ----
        v16h pa0 = ld_frag_cont(&pw[row * 64 + bb]);
        v16h pa1 = ld_frag_cont(&pw[row * 64 + 32 + bb]);
        #pragma unroll
        for (int dt = 0; dt < 4; ++dt) {
            const _Float16* vr = vl + (dt * 16 + row) * 64;
            oacc[dt] = wmma_f32_f16(pa0, ld_frag_cont(vr + bb),      oacc[dt]);
            oacc[dt] = wmma_f32_f16(pa1, ld_frag_cont(vr + 32 + bb), oacc[dt]);
        }
    }

    // normalize, store ctx row-major [4096][1280] f16
    const int hh = bh % 20, bo_ = bh / 20;
    #pragma unroll
    for (int r = 0; r < 8; ++r) {
        float inv = 1.f / lrun[r];
        int l = q0 + r + ((lane < 16) ? 0 : 8);
        #pragma unroll
        for (int dt = 0; dt < 4; ++dt) {
            int col = hh * 64 + dt * 16 + (lane & 15);
            ctxh[((size_t)(bo_ * 1024) + l) * 1280 + col] =
                (_Float16)(oacc[dt][r] * inv);
        }
    }
}

// ---------------------------------------------------------------------------
// Kernel 4: output projection ctx @ Wo + bo -> fp32 out.  128x128 tiles.
// ---------------------------------------------------------------------------
__global__ __launch_bounds__(256) void out_gemm(
    const _Float16* __restrict__ A, const float* __restrict__ Wo,
    const float* __restrict__ bo, float* __restrict__ out)
{
    __shared__ __align__(16) _Float16 At[128 * 32];
    __shared__ __align__(16) _Float16 Bt[8][16 * 32];

    const int tid  = threadIdx.x;
    const int wave = tid >> 5;
    const int lane = tid & 31;
    const int m0 = blockIdx.x * 128;
    const int n0 = blockIdx.y * 128;

    const int row = lane & 15;
    const int b1  = (lane < 16) ? 0 : 8;
    const int bb  = (lane < 16) ? 0 : 16;

    v8f acc[8] = {};
    for (int kk = 0; kk < 1280; kk += 32) {
        if (kk + 32 < 1280) {
            __builtin_prefetch(&A[(m0 + (tid >> 1)) * 1280 + kk + 32], 0, 0);
            __builtin_prefetch(&Wo[(kk + 32 + (tid >> 3)) * 1280 + n0], 0, 0);
        }
        #pragma unroll
        for (int i = 0; i < 2; ++i) {
            int idx8 = tid + i * 256;
            int r = idx8 >> 2;
            int c = (idx8 & 3) * 8;
            *(v8h*)&At[r * 32 + c] = *(const v8h*)&A[(m0 + r) * 1280 + kk + c];
        }
        #pragma unroll
        for (int i = 0; i < 2; ++i) {
            int idx = tid + i * 256;
            int kp = idx >> 5;
            int c  = (idx & 31) * 4;
            v4f x0 = *(const v4f*)&Wo[(kk + 2 * kp    ) * 1280 + n0 + c];
            v4f x1 = *(const v4f*)&Wo[(kk + 2 * kp + 1) * 1280 + n0 + c];
            #pragma unroll
            for (int u = 0; u < 4; ++u) {
                int cc = c + u;
                v2h h; h.x = (_Float16)x0[u]; h.y = (_Float16)x1[u];
                *(v2h*)&Bt[cc >> 4][(cc & 15) * 32 + 2 * kp] = h;
            }
        }
        __syncthreads();
        v16h a = ld_frag_cont(&At[(wave * 16 + row) * 32 + bb]);
        #pragma unroll
        for (int nt = 0; nt < 8; ++nt) {
            v16h b = ld_frag_cont(&Bt[nt][row * 32 + bb]);
            acc[nt] = wmma_f32_f16(a, b, acc[nt]);
        }
        __syncthreads();
    }

    #pragma unroll
    for (int nt = 0; nt < 8; ++nt) {
        #pragma unroll
        for (int r = 0; r < 8; ++r) {
            int m = m0 + wave * 16 + r + ((lane < 16) ? 0 : 8);
            int n = n0 + nt * 16 + (lane & 15);
            out[m * 1280 + n] = acc[nt][r] + bo[n];
        }
    }
}

// ---------------------------------------------------------------------------
extern "C" void kernel_launch(void* const* d_in, const int* in_sizes, int n_in,
                              void* d_out, int out_size, void* d_ws, size_t ws_size,
                              hipStream_t stream) {
    const float* X   = (const float*)d_in[0];
    const float* Wq  = (const float*)d_in[1];
    const float* Wk  = (const float*)d_in[2];
    const float* Wv  = (const float*)d_in[3];
    const float* Wo  = (const float*)d_in[4];
    const float* bo  = (const float*)d_in[5];
    const float* Kbg = (const float*)d_in[6];
    const float* Vbg = (const float*)d_in[7];
    float* out = (float*)d_out;

    // workspace (f16): Qs 10MB | Kc 12.5MB | Vt 12.5MB | ctxh 10MB  (~45 MB)
    char* ws = (char*)d_ws;
    _Float16* Qs   = (_Float16*)ws;  ws += (size_t)80 * 1024 * 64 * 2;
    _Float16* Kc   = (_Float16*)ws;  ws += (size_t)80 * 1280 * 64 * 2;
    _Float16* Vt   = (_Float16*)ws;  ws += (size_t)80 * 64 * 1280 * 2;
    _Float16* ctxh = (_Float16*)ws;

    qkv_gemm<<<dim3(32, 10, 3), 256, 0, stream>>>(X, Wq, Wk, Wv, Qs, Kc, Vt);
    bank_pool<<<dim3(2560), 256, 0, stream>>>(Kbg, Vbg, Kc, Vt);
    attn<<<dim3(8, 80), 256, 0, stream>>>(Qs, Kc, Vt, ctxh);
    out_gemm<<<dim3(32, 10), 256, 0, stream>>>(ctxh, Wo, bo, out);
}
